// MultiHeadAttention_37495064494145
// MI455X (gfx1250) — matrix operand order repaired and checked by reading, not compiled, not run on hardware
//
#include <hip/hip_runtime.h>
#include <hip/hip_bf16.h>
#include <math.h>

typedef __bf16 bf16;
typedef __attribute__((ext_vector_type(16))) __bf16 v16bf;
typedef __attribute__((ext_vector_type(8)))  __bf16 v8bf;
typedef __attribute__((ext_vector_type(8)))  float  v8f;

#define SEQ     2048
#define DMODEL  1024
#define NHEADS  16
#define HDIM    64

__device__ __forceinline__ v8f wmma_bf16(v16bf a, v16bf b, v8f c) {
    return __builtin_amdgcn_wmma_f32_16x16x32_bf16(false, a, false, b, (short)0, c, false, false);
}

__device__ __forceinline__ v16bf bf16x16_frag(v8bf lo, v8bf hi) {
    v16bf r;
#pragma unroll
    for (int j = 0; j < 8; ++j) { r[j] = lo[j]; r[8 + j] = hi[j]; }
    return r;
}

// A-operand fragment: row `row`, 8-elem chunks at k0+8*half and k0+16+8*half
__device__ __forceinline__ v16bf load_afrag(const bf16* row, int k0, int half) {
    const bf16* p = row + k0 + half * 8;
    return bf16x16_frag(*(const v8bf*)p, *(const v8bf*)(p + 16));
}
// B-operand fragment: row n, 8-elem chunks at k0+8*half and k0+16+8*half
__device__ __forceinline__ v16bf load_bfrag(const bf16* nrow, int k0, int half) {
    const bf16* p = nrow + k0 + half * 8;
    return bf16x16_frag(*(const v8bf*)p, *(const v8bf*)(p + 16));
}

// ---------------------------------------------------------------------------
// Kernel 0: bulk f32 -> bf16 conversion (bandwidth pass, done once per call)
// ---------------------------------------------------------------------------
__global__ __launch_bounds__(256)
void cvt_f32_bf16(const float* __restrict__ in, bf16* __restrict__ out, int n)
{
    int i = (blockIdx.x * 256 + threadIdx.x) * 8;
    if (i + 8 <= n) {
        float4 f0 = *(const float4*)(in + i);
        float4 f1 = *(const float4*)(in + i + 4);
        v8bf pk;
        pk[0] = (bf16)f0.x; pk[1] = (bf16)f0.y; pk[2] = (bf16)f0.z; pk[3] = (bf16)f0.w;
        pk[4] = (bf16)f1.x; pk[5] = (bf16)f1.y; pk[6] = (bf16)f1.z; pk[7] = (bf16)f1.w;
        *(v8bf*)(out + i) = pk;
    }
}

// ---------------------------------------------------------------------------
// Kernel A: QKV projection (Ebf @ Wbf^T) with fused RoPE on Q/K.
// grid: (16 m-blocks, 16 heads, 3 matrices), block 256 (8 waves).
// Each wave: 32 rows (s) x 64 cols (one head) = 8 accumulators; each B frag
// feeds two WMMAs.
// ---------------------------------------------------------------------------
__global__ __launch_bounds__(256)
void qkv_rope(const bf16* __restrict__ E,
              const bf16* __restrict__ Wq, const bf16* __restrict__ Wk,
              const bf16* __restrict__ Wv,
              bf16* __restrict__ Q, bf16* __restrict__ K, bf16* __restrict__ Vt)
{
    const int lane = threadIdx.x & 31;
    const int wave = threadIdx.x >> 5;
    const int half = lane >> 4;
    const int l16  = lane & 15;
    const int m0   = blockIdx.x * 256 + wave * 32;   // global row (b*SEQ + s)
    const int h    = blockIdx.y;
    const int mat  = blockIdx.z;                     // 0=Q 1=K 2=V
    const bf16* W = (mat == 0) ? Wq : (mat == 1) ? Wk : Wv;

    v8f acc[2][4] = {};
    const bf16* Arow0 = E + (size_t)(m0 + l16) * DMODEL;
    const bf16* Arow1 = Arow0 + (size_t)16 * DMODEL;
    const bf16* Wh    = W + (size_t)(h * HDIM + l16) * DMODEL;

    for (int k0 = 0; k0 < DMODEL; k0 += 32) {
        v16bf a0 = load_afrag(Arow0, k0, half);
        v16bf a1 = load_afrag(Arow1, k0, half);
#pragma unroll
        for (int t = 0; t < 4; ++t) {
            v16bf bfr = load_bfrag(Wh + (size_t)(t * 16) * DMODEL, k0, half);
            acc[0][t] = wmma_bf16(a0, bfr, acc[0][t]);
            acc[1][t] = wmma_bf16(a1, bfr, acc[1][t]);
        }
    }

    const int b = m0 / SEQ;
#pragma unroll
    for (int mi = 0; mi < 2; ++mi) {
        const int s0 = (m0 & (SEQ - 1)) + mi * 16;
        if (mat < 2) {
            // RoPE: partner dim d^32 lives in accumulator t^2, same lane
            v8f rot[4];
#pragma unroll
            for (int t = 0; t < 4; ++t) {
                const int d    = t * 16 + l16;
                const float sg = (t < 2) ? -1.0f : 1.0f;
                const int  tp  = t ^ 2;
                const float invf = __expf(-0.28782313662425572f * (float)(d & 31));
#pragma unroll
                for (int r = 0; r < 8; ++r) {
                    float s = (float)(s0 + r + 8 * half);
                    float c, sn; __sincosf(s * invf, &c, &sn);
                    rot[t][r] = acc[mi][t][r] * c + sg * acc[mi][tp][r] * sn;
                }
            }
            bf16* out = (mat == 0) ? Q : K;
            size_t base = ((size_t)(b * NHEADS + h) * SEQ + s0) * HDIM;
#pragma unroll
            for (int t = 0; t < 4; ++t) {
                const int d = t * 16 + l16;
#pragma unroll
                for (int r = 0; r < 8; ++r)
                    out[base + (size_t)(r + 8 * half) * HDIM + d] = (bf16)rot[t][r];
            }
        } else {
            // V transposed: Vt[b][h][d][s]; 8 contiguous s per lane -> b128 store
#pragma unroll
            for (int t = 0; t < 4; ++t) {
                const int d = t * 16 + l16;
                bf16* vp = Vt + ((size_t)(b * NHEADS + h) * HDIM + d) * SEQ + s0 + 8 * half;
                v8bf pk;
#pragma unroll
                for (int r = 0; r < 8; ++r) pk[r] = (bf16)acc[mi][t][r];
                *(v8bf*)vp = pk;
            }
        }
    }
}

// ---------------------------------------------------------------------------
// Kernel B: causal flash attention. grid (16 q-blocks, 16 heads, 2 batch),
// block 256 = 8 waves; one wave owns a 16-row q-tile, k in chunks of 32.
// ---------------------------------------------------------------------------
__global__ __launch_bounds__(256)
void attn(const bf16* __restrict__ Q, const bf16* __restrict__ K,
          const bf16* __restrict__ Vt, bf16* __restrict__ Ab)
{
    __shared__ __align__(16) bf16 lds[8][16 * 32];   // per-wave P transpose tile

    const int lane = threadIdx.x & 31;
    const int wave = threadIdx.x >> 5;
    const int half = lane >> 4;
    const int l16  = lane & 15;
    const int q0   = blockIdx.x * 128 + wave * 16;
    const int h    = blockIdx.y;
    const int b    = blockIdx.z;

    const bf16* Qh = Q  + (size_t)(b * NHEADS + h) * SEQ * HDIM;
    const bf16* Kh = K  + (size_t)(b * NHEADS + h) * SEQ * HDIM;
    const bf16* Vh = Vt + (size_t)(b * NHEADS + h) * HDIM * SEQ;

    v16bf qa[2];
#pragma unroll
    for (int dd = 0; dd < 2; ++dd) {
        const bf16* p = Qh + (size_t)(q0 + l16) * HDIM + dd * 32 + half * 8;
        qa[dd] = bf16x16_frag(*(const v8bf*)p, *(const v8bf*)(p + 16));
    }

    float M[8], L[8];
    v8f o[4] = {};
#pragma unroll
    for (int r = 0; r < 8; ++r) { M[r] = -1e30f; L[r] = 0.0f; }
    const float scale = 0.125f;   // 1/sqrt(64)
    const int kend = q0 + 16;
    bf16* P = lds[wave];

    for (int kc = 0; kc < kend; kc += 32) {
        __builtin_prefetch(Kh + (size_t)(kc + 32 + l16) * HDIM, 0, 1);
        v8f s0v = {}, s1v = {};
#pragma unroll
        for (int dd = 0; dd < 2; ++dd) {
            v16bf kb0 = load_bfrag(Kh + (size_t)(kc + l16) * HDIM,      dd * 32, half);
            s0v = wmma_bf16(qa[dd], kb0, s0v);
            v16bf kb1 = load_bfrag(Kh + (size_t)(kc + 16 + l16) * HDIM, dd * 32, half);
            s1v = wmma_bf16(qa[dd], kb1, s1v);
        }
        float p0s[8], p1s[8];
#pragma unroll
        for (int r = 0; r < 8; ++r) {
            const int qrow = q0 + r + 8 * half;
            float x0 = (kc + l16      <= qrow) ? s0v[r] * scale : -1e30f;
            float x1 = (kc + 16 + l16 <= qrow) ? s1v[r] * scale : -1e30f;
            float mx = fmaxf(x0, x1);
#pragma unroll
            for (int ofs = 1; ofs < 16; ofs <<= 1) mx = fmaxf(mx, __shfl_xor(mx, ofs, 32));
            float mnew  = fmaxf(M[r], mx);
            float alpha = __expf(M[r] - mnew);
            M[r] = mnew;
            float e0 = __expf(x0 - mnew);
            float e1 = __expf(x1 - mnew);
            float rs = e0 + e1;
#pragma unroll
            for (int ofs = 1; ofs < 16; ofs <<= 1) rs += __shfl_xor(rs, ofs, 32);
            L[r] = L[r] * alpha + rs;
#pragma unroll
            for (int t = 0; t < 4; ++t) o[t][r] *= alpha;
            p0s[r] = e0; p1s[r] = e1;
        }
        // transpose P (C-layout -> A-layout) through per-wave LDS
#pragma unroll
        for (int r = 0; r < 8; ++r) {
            const int row = r + 8 * half;
            P[row * 32 + l16]      = (bf16)p0s[r];
            P[row * 32 + 16 + l16] = (bf16)p1s[r];
        }
        asm volatile("s_wait_dscnt 0x0" ::: "memory");
        const bf16* pp = P + l16 * 32 + half * 8;
        v16bf pa = bf16x16_frag(*(const v8bf*)pp, *(const v8bf*)(pp + 16));
#pragma unroll
        for (int t = 0; t < 4; ++t) {
            v16bf vb = load_bfrag(Vh + (size_t)(t * 16 + l16) * SEQ, kc, half);
            o[t] = wmma_bf16(pa, vb, o[t]);
        }
        asm volatile("s_wait_dscnt 0x0" ::: "memory");
    }

    const size_t mrow = (size_t)b * SEQ + q0;
#pragma unroll
    for (int t = 0; t < 4; ++t) {
        const int d = h * HDIM + t * 16 + l16;
#pragma unroll
        for (int r = 0; r < 8; ++r) {
            Ab[(mrow + r + 8 * half) * DMODEL + d] = (bf16)(o[t][r] / L[r]);
        }
    }
}

// ---------------------------------------------------------------------------
// Kernel C: output projection  Out = Ab(bf16) @ Wo_bf^T + bo   (fp32 out)
// grid (16 m-blocks, 16 n-strips), 32x64 per wave.
// ---------------------------------------------------------------------------
__global__ __launch_bounds__(256)
void out_proj(const bf16* __restrict__ Ab, const bf16* __restrict__ Wo,
              const float* __restrict__ bo, float* __restrict__ Out)
{
    const int lane = threadIdx.x & 31;
    const int wave = threadIdx.x >> 5;
    const int half = lane >> 4;
    const int l16  = lane & 15;
    const int m0   = blockIdx.x * 256 + wave * 32;
    const int n0   = blockIdx.y * 64;

    v8f acc[2][4] = {};
    const bf16* Arow0 = Ab + (size_t)(m0 + l16) * DMODEL;
    const bf16* Arow1 = Arow0 + (size_t)16 * DMODEL;
    const bf16* Wn    = Wo + (size_t)(n0 + l16) * DMODEL;

    for (int k0 = 0; k0 < DMODEL; k0 += 32) {
        v16bf a0 = load_afrag(Arow0, k0, half);
        v16bf a1 = load_afrag(Arow1, k0, half);
#pragma unroll
        for (int t = 0; t < 4; ++t) {
            v16bf bfr = load_bfrag(Wn + (size_t)(t * 16) * DMODEL, k0, half);
            acc[0][t] = wmma_bf16(a0, bfr, acc[0][t]);
            acc[1][t] = wmma_bf16(a1, bfr, acc[1][t]);
        }
    }
#pragma unroll
    for (int mi = 0; mi < 2; ++mi) {
#pragma unroll
        for (int t = 0; t < 4; ++t) {
            const int n = n0 + t * 16 + l16;
            const float bias = bo[n];
#pragma unroll
            for (int r = 0; r < 8; ++r) {
                const int m = m0 + mi * 16 + r + 8 * half;
                Out[(size_t)m * DMODEL + n] = acc[mi][t][r] + bias;
            }
        }
    }
}

// ---------------------------------------------------------------------------
extern "C" void kernel_launch(void* const* d_in, const int* in_sizes, int n_in,
                              void* d_out, int out_size, void* d_ws, size_t ws_size,
                              hipStream_t stream)
{
    const float* E  = (const float*)d_in[0];
    const float* Wq = (const float*)d_in[1];
    const float* Wk = (const float*)d_in[2];
    const float* Wv = (const float*)d_in[3];
    const float* Wo = (const float*)d_in[4];
    const float* bo = (const float*)d_in[5];
    float* Out = (float*)d_out;

    char* ws = (char*)d_ws;
    const size_t SZ_QKV = (size_t)2 * NHEADS * SEQ * HDIM * sizeof(bf16);  // 8 MB
    const size_t SZ_W   = (size_t)DMODEL * DMODEL * sizeof(bf16);          // 2 MB
    bf16* Qb  = (bf16*)(ws);
    bf16* Kb  = (bf16*)(ws + SZ_QKV);
    bf16* Vt  = (bf16*)(ws + 2 * SZ_QKV);
    bf16* Ab  = (bf16*)(ws + 3 * SZ_QKV);
    bf16* Ebf = (bf16*)(ws + 4 * SZ_QKV);
    bf16* Wqb = (bf16*)(ws + 5 * SZ_QKV);
    bf16* Wkb = (bf16*)(ws + 5 * SZ_QKV + SZ_W);
    bf16* Wvb = (bf16*)(ws + 5 * SZ_QKV + 2 * SZ_W);
    bf16* Wob = (bf16*)(ws + 5 * SZ_QKV + 3 * SZ_W);

    const int nE = 2 * SEQ * DMODEL;     // 4 Mi elems
    const int nW = DMODEL * DMODEL;      // 1 Mi elems
    cvt_f32_bf16<<<nE / 2048, 256, 0, stream>>>(E,  Ebf, nE);
    cvt_f32_bf16<<<nW / 2048, 256, 0, stream>>>(Wq, Wqb, nW);
    cvt_f32_bf16<<<nW / 2048, 256, 0, stream>>>(Wk, Wkb, nW);
    cvt_f32_bf16<<<nW / 2048, 256, 0, stream>>>(Wv, Wvb, nW);
    cvt_f32_bf16<<<nW / 2048, 256, 0, stream>>>(Wo, Wob, nW);

    dim3 gA(16, NHEADS, 3);
    qkv_rope<<<gA, 256, 0, stream>>>(Ebf, Wqb, Wkb, Wvb, Qb, Kb, Vt);

    dim3 gB(16, NHEADS, 2);
    attn<<<gB, 256, 0, stream>>>(Qb, Kb, Vt, Ab);

    dim3 gC(16, NHEADS, 1);
    out_proj<<<gC, 256, 0, stream>>>(Ab, Wob, bo, Out);
}
